// DistributedMoELayer_77567109366043
// MI455X (gfx1250) — matrix operand order repaired and checked by reading, not compiled, not run on hardware
//
#include <hip/hip_runtime.h>

#define HID 2048
#define FFN 8192
#define NE 8
#define NT 8192          // 4*2048 tokens
#define BM 32            // tokens per block
#define FC 256           // FFN chunk per outer iteration

typedef __attribute__((ext_vector_type(16))) __bf16 v16bf;
typedef __attribute__((ext_vector_type(8)))  float  v8f;
typedef __attribute__((ext_vector_type(4)))  float  f32x4;
typedef __attribute__((ext_vector_type(4)))  unsigned short u16x4;

// LDS strides (bf16 elements); byte strides are multiples of 16
#define SX_STRIDE 2056   // [32][2056]  full X block rows (2048 valid k)
#define SH_STRIDE 264    // [32][264]   H tile (256 valid cols)

__device__ __forceinline__ float gelu_exact(float x) {
    return 0.5f * x * (1.0f + erff(x * 0.70710678118654752440f));
}
__device__ __forceinline__ unsigned short f2bf_bits(float f) {
    __bf16 b = (__bf16)f;                    // RTE
    return __builtin_bit_cast(unsigned short, b);
}

// ---- CDNA5 async global->LDS copy (16B per lane), tracked on ASYNCcnt ----
__device__ __forceinline__ void async_b128(unsigned short* lds_dst, const unsigned short* gsrc) {
    unsigned ldsa = (unsigned)(size_t)lds_dst;   // flat LDS addr[31:0] = LDS offset
    asm volatile("global_load_async_to_lds_b128 %0, %1, off"
                 :: "v"(ldsa), "v"(gsrc) : "memory");
}
__device__ __forceinline__ void wait_async0() {
    asm volatile("s_wait_asynccnt 0x0" ::: "memory");
}

// A fragment (16x32 bf16): lane<16 -> row=lane, K {0..7,16..23}; lane>=16 -> row=lane-16, K {8..15,24..31}
__device__ __forceinline__ v16bf frag_a(const unsigned short* rowp, int kw, int lane) {
    int kb = (lane & 16) ? 8 : 0;
    v16bf f;
    *((f32x4*)&f)     = *(const f32x4*)(rowp + kw + kb);
    *(((f32x4*)&f)+1) = *(const f32x4*)(rowp + kw + kb + 16);
    return f;
}
// B fragment (32x16 bf16): lane<16 -> col=lane, K {0..7,16..23}; lane>=16 -> col=lane-16, K {8..15,24..31}
__device__ __forceinline__ v16bf frag_b(const unsigned short* rowp, int kw, int lane) {
    int kb = (lane & 16) ? 8 : 0;
    v16bf f;
    *((f32x4*)&f)     = *(const f32x4*)(rowp + kw + kb);
    *(((f32x4*)&f)+1) = *(const f32x4*)(rowp + kw + kb + 16);
    return f;
}

// ---------------- Kernel 0a: tiled fp32 -> bf16 transpose ----------------
// src: [E][R][C] f32   dst: [E][C][R] bf16
__global__ void transpose_bf16_kernel(const float* __restrict__ src,
                                      unsigned short* __restrict__ dst,
                                      int R, int C, int tilesR, int tilesC) {
    __shared__ float tile[32][33];
    int b = blockIdx.x;
    int e = b / (tilesR * tilesC);
    int rem = b % (tilesR * tilesC);
    int tr = rem / tilesC, tc = rem % tilesC;

    int hl = threadIdx.x >> 3, fl = (threadIdx.x & 7) * 4;
    f32x4 v = *(const f32x4*)(src + ((size_t)e * R + tr * 32 + hl) * C + tc * 32 + fl);
    tile[hl][fl] = v.x; tile[hl][fl + 1] = v.y; tile[hl][fl + 2] = v.z; tile[hl][fl + 3] = v.w;
    __syncthreads();

    int cl = threadIdx.x >> 3, rl = (threadIdx.x & 7) * 4;
    u16x4 o;
    o.x = f2bf_bits(tile[rl][cl]);
    o.y = f2bf_bits(tile[rl + 1][cl]);
    o.z = f2bf_bits(tile[rl + 2][cl]);
    o.w = f2bf_bits(tile[rl + 3][cl]);
    *(u16x4*)(dst + ((size_t)e * C + tc * 32 + cl) * R + tr * 32 + rl) = o;
}

// ---------------- Kernel 0b: x fp32 -> bf16 ----------------
__global__ void cvt_x_kernel(const float* __restrict__ x, unsigned short* __restrict__ xbf) {
    size_t i = (size_t)blockIdx.x * blockDim.x + threadIdx.x;   // float4 index
    f32x4 v = ((const f32x4*)x)[i];
    u16x4 o;
    o.x = f2bf_bits(v.x); o.y = f2bf_bits(v.y); o.z = f2bf_bits(v.z); o.w = f2bf_bits(v.w);
    ((u16x4*)xbf)[i] = o;
}

// ---------------- Kernel 1: router (one wave per token) ----------------
__global__ void moe_router_kernel(const float* __restrict__ x,
                                  const float* __restrict__ gw,
                                  int* __restrict__ topk_idx,
                                  float* __restrict__ topk_w) {
    int t = blockIdx.x * 8 + (threadIdx.x >> 5);
    int lane = threadIdx.x & 31;
    const float* xr = x + (size_t)t * HID;
    float acc[NE];
#pragma unroll
    for (int e = 0; e < NE; ++e) acc[e] = 0.f;
    for (int h = lane; h < HID; h += 32) {
        float xv = xr[h];
#pragma unroll
        for (int e = 0; e < NE; ++e) acc[e] += xv * gw[e * HID + h];
    }
#pragma unroll
    for (int e = 0; e < NE; ++e)
#pragma unroll
        for (int off = 16; off > 0; off >>= 1) acc[e] += __shfl_xor(acc[e], off, 32);
    if (lane == 0) {
        float mx = acc[0];
#pragma unroll
        for (int e = 1; e < NE; ++e) mx = fmaxf(mx, acc[e]);
        float p[NE];
#pragma unroll
        for (int e = 0; e < NE; ++e) p[e] = __expf(acc[e] - mx);
        int i1 = 0;
#pragma unroll
        for (int e = 1; e < NE; ++e) if (p[e] > p[i1]) i1 = e;
        int i2 = (i1 == 0) ? 1 : 0;
#pragma unroll
        for (int e = 0; e < NE; ++e) if (e != i1 && p[e] > p[i2]) i2 = e;
        float tot = p[i1] + p[i2];
        topk_idx[t * 2] = i1;  topk_idx[t * 2 + 1] = i2;
        topk_w[t * 2] = p[i1] / tot;  topk_w[t * 2 + 1] = p[i2] / tot;
    }
}

// ---------------- Kernel 2: deterministic per-expert list build ----------------
__global__ void moe_build_lists_kernel(const int* __restrict__ topk_idx,
                                       const float* __restrict__ topk_w,
                                       int* __restrict__ list_idx,
                                       float* __restrict__ list_w,
                                       int* __restrict__ list_slot,
                                       int* __restrict__ counts) {
    int e = threadIdx.x >> 5;           // 8 waves, one expert each
    int lane = threadIdx.x & 31;
    int cnt = 0;
    for (int base = 0; base < NT; base += 32) {
        int t = base + lane;
        int i0 = topk_idx[t * 2], i1 = topk_idx[t * 2 + 1];
        int slot = (i0 == e) ? 0 : ((i1 == e) ? 1 : -1);
        unsigned long long bal = __ballot(slot >= 0);
        int pre = __popcll(bal & ((1ull << lane) - 1ull));
        if (slot >= 0) {
            int pos = cnt + pre;
            list_idx[e * NT + pos] = t;
            list_w[e * NT + pos] = topk_w[t * 2 + slot];
            list_slot[e * NT + pos] = slot;
        }
        cnt += __popcll(bal);
    }
    if (lane == 0) counts[e] = cnt;
}

// ---------------- Kernel 3: fused grouped FFN (WMMA, weights direct-from-global) ----
__global__ void __launch_bounds__(512)
moe_ffn_kernel(const unsigned short* __restrict__ xbf,   // [NT][H] bf16
               const unsigned short* __restrict__ w1t,   // [E][F][H] bf16
               const float* __restrict__ b1,
               const unsigned short* __restrict__ w2t,   // [E][H][F] bf16
               const float* __restrict__ b2,
               const int* __restrict__ list_idx, const float* __restrict__ list_w,
               const int* __restrict__ list_slot, const int* __restrict__ counts,
               float* __restrict__ contrib) {
    extern __shared__ unsigned char smem_raw[];
    unsigned short* sX = (unsigned short*)smem_raw;      // [BM][SX_STRIDE] resident X block
    unsigned short* sH = sX + BM * SX_STRIDE;            // [BM][SH_STRIDE] H tile
    int*   sTok  = (int*)(sH + BM * SH_STRIDE);
    float* sW    = (float*)(sTok + BM);
    int*   sSlot = (int*)(sW + BM);

    const int e   = blockIdx.x >> 8;                     // 256 blocks per expert
    const int blk = blockIdx.x & 255;
    const int cnt = counts[e];
    if (blk * BM >= cnt) return;

    const int tid   = threadIdx.x;
    const int lane  = tid & 31;
    const int wave  = tid >> 5;                          // 0..15
    const int mlane = lane & 15;
    const int hi8   = (lane & 16) ? 8 : 0;

    if (tid < BM) {
        int p = blk * BM + tid;
        if (p < cnt) {
            sTok[tid]  = list_idx[e * NT + p];
            sW[tid]    = list_w[e * NT + p];
            sSlot[tid] = list_slot[e * NT + p];
        } else { sTok[tid] = 0; sW[tid] = 0.f; sSlot[tid] = 0; }
    }
    __syncthreads();

    // ---- load gathered X block into LDS once (async DMA, 128 KB) ----
    {
        int m = tid >> 4;
        const unsigned short* src = xbf + (size_t)sTok[m] * HID;
        unsigned short* dst = sX + m * SX_STRIDE;
#pragma unroll
        for (int j = 0; j < 16; ++j) {
            int seg = (tid & 15) + j * 16;               // 16 x 8 elems = 128 per pass
            async_b128(dst + seg * 8, src + seg * 8);
        }
        wait_async0();
    }
    __syncthreads();

    const size_t w1te = (size_t)e * FFN * HID;           // [F][H]
    const size_t w2te = (size_t)e * HID * FFN;           // [H][F]

    // ---- fire-and-forget prefetch of one f-chunk's weights into near cache ----
    // W1^T: 16 rows (this wave's f-cols) x 32 cachelines; W2^T: 128 rows x 4 lines.
    // One global_prefetch covers 32 lanes x 128B = 4KB -> 16+16 instrs per chunk.
    auto prefetch_chunk = [&](int fcp) {
        const unsigned short* p1 = w1t + w1te +
            (size_t)(fcp + wave * 16 + mlane) * HID + (lane >> 4) * 64;
#pragma unroll
        for (int j = 0; j < 16; ++j)
            __builtin_prefetch(p1 + j * 128, 0, 1);      // lines {h, h+2, ..., h+30} per row
        const unsigned short* p2 = w2t + w2te +
            (size_t)(wave * 128 + (lane >> 2)) * FFN + fcp + (lane & 3) * 64;
#pragma unroll
        for (int j = 0; j < 16; ++j)
            __builtin_prefetch(p2 + (size_t)j * 8 * FFN, 0, 1);  // rows +{0,8,...,120}
    };
    prefetch_chunk(0);

    const v8f vzero = {0.f, 0.f, 0.f, 0.f, 0.f, 0.f, 0.f, 0.f};
    v8f acc2[16];                                        // out^T: [8 n-tiles][2 m-tiles]
#pragma unroll
    for (int j = 0; j < 16; ++j) acc2[j] = vzero;

    const unsigned short* arow0 = sX + mlane * SX_STRIDE;        // m-tile 0 rows
    const unsigned short* arow1 = sX + (16 + mlane) * SX_STRIDE; // m-tile 1 rows

    for (int fc = 0; fc < FFN; fc += FC) {
        // ---- GEMM1: wave owns f-cols [fc+wave*16, +16), both m-tiles; B from global w1t.
        //      B fragments are prefetched one iteration ahead (the final prefetch reads
        //      into the adjacent w1t row / following ws region: benign, unused). ----
        v8f accA0 = vzero, accB0 = vzero, accA1 = vzero, accB1 = vzero;
        const unsigned short* browg = w1t + w1te + (size_t)(fc + wave * 16 + mlane) * HID;
        v16bf b0  = frag_b(browg, 0,  lane);
        v16bf b1f = frag_b(browg, 32, lane);
#pragma unroll 4
        for (int kw = 0; kw < HID; kw += 64) {
            v16bf nb0  = frag_b(browg, kw + 64, lane);
            v16bf nb1f = frag_b(browg, kw + 96, lane);
            v16bf a00 = frag_a(arow0, kw,      lane);
            v16bf a01 = frag_a(arow0, kw + 32, lane);
            v16bf a10 = frag_a(arow1, kw,      lane);
            v16bf a11 = frag_a(arow1, kw + 32, lane);
            accA0 = __builtin_amdgcn_wmma_f32_16x16x32_bf16(false, a00, false, b0,  (short)0, accA0, false, false);
            accB0 = __builtin_amdgcn_wmma_f32_16x16x32_bf16(false, a01, false, b1f, (short)0, accB0, false, false);
            accA1 = __builtin_amdgcn_wmma_f32_16x16x32_bf16(false, a10, false, b0,  (short)0, accA1, false, false);
            accB1 = __builtin_amdgcn_wmma_f32_16x16x32_bf16(false, a11, false, b1f, (short)0, accB1, false, false);
            b0 = nb0; b1f = nb1f;
        }
        v8f h0 = accA0 + accB0;
        v8f h1 = accA1 + accB1;

        // prefetch next chunk's weights while this chunk's GEMM2 runs
        prefetch_chunk(fc + FC < FFN ? fc + FC : fc);

        __syncthreads();          // all waves finished reading sH of previous chunk
        {
            int fcol = wave * 16 + mlane;
            float bias = b1[(size_t)e * FFN + fc + fcol];
#pragma unroll
            for (int r = 0; r < 8; ++r) {
                sH[(hi8 + r) * SH_STRIDE + fcol]        = f2bf_bits(gelu_exact(h0[r] + bias));
                sH[(16 + hi8 + r) * SH_STRIDE + fcol]   = f2bf_bits(gelu_exact(h1[r] + bias));
            }
        }
        __syncthreads();          // sH of this chunk visible to all waves

        // ---- GEMM2 (transposed): out^T[n][m] += W2^T[n][k] * H^T[k][m]. ----
        for (int ks = 0; ks < FC; ks += 32) {
            v16bf bm0 = frag_b(sH + mlane * SH_STRIDE,        ks, lane);
            v16bf bm1 = frag_b(sH + (16 + mlane) * SH_STRIDE, ks, lane);
            const unsigned short* abase =
                w2t + w2te + (size_t)(wave * 128 + mlane) * FFN + fc + ks;
            v16bf aw[8];
#pragma unroll
            for (int jt = 0; jt < 8; ++jt)
                aw[jt] = frag_a(abase + (size_t)jt * 16 * FFN, 0, lane);
#pragma unroll
            for (int jt = 0; jt < 8; ++jt) {
                acc2[jt * 2 + 0] = __builtin_amdgcn_wmma_f32_16x16x32_bf16(false, aw[jt], false, bm0,
                                                                           (short)0, acc2[jt * 2 + 0], false, false);
                acc2[jt * 2 + 1] = __builtin_amdgcn_wmma_f32_16x16x32_bf16(false, aw[jt], false, bm1,
                                                                           (short)0, acc2[jt * 2 + 1], false, false);
            }
        }
    }

    // ---- epilogue: contrib[(t,slot)][n] = gate_w * (acc + b2[n]) ----
    {
#pragma unroll
        for (int jt = 0; jt < 8; ++jt) {
#pragma unroll
            for (int mt = 0; mt < 2; ++mt) {
                int m = mt * 16 + mlane;                 // token row (C layout: lane = column)
                int p = blk * BM + m;
                if (p < cnt) {
                    int   t    = sTok[m];
                    int   slot = sSlot[m];
                    float gwt  = sW[m];
                    float* orow = contrib + ((size_t)t * 2 + slot) * HID;
#pragma unroll
                    for (int r = 0; r < 8; ++r) {
                        int n = wave * 128 + jt * 16 + hi8 + r;
                        orow[n] = (acc2[jt * 2 + mt][r] + b2[(size_t)e * HID + n]) * gwt;
                    }
                }
            }
        }
    }
}

// ---------------- Kernel 4: deterministic combine ----------------
__global__ void moe_combine_kernel(const float* __restrict__ x,
                                   const float* __restrict__ contrib,
                                   float* __restrict__ out) {
    size_t i = (size_t)blockIdx.x * blockDim.x + threadIdx.x;  // float4 index
    size_t total = (size_t)NT * HID / 4;
    if (i >= total) return;
    size_t t  = i >> 9;                // 512 float4 per token
    size_t n4 = i & 511;
    f32x4 xv = ((const f32x4*)x)[i];
    f32x4 c0 = ((const f32x4*)(contrib + (t * 2) * HID))[n4];
    f32x4 c1 = ((const f32x4*)(contrib + (t * 2 + 1) * HID))[n4];
    ((f32x4*)out)[i] = xv + c0 + c1;
}

extern "C" void kernel_launch(void* const* d_in, const int* in_sizes, int n_in,
                              void* d_out, int out_size, void* d_ws, size_t ws_size,
                              hipStream_t stream) {
    const float* x      = (const float*)d_in[0];
    const float* gate_w = (const float*)d_in[1];
    const float* w1     = (const float*)d_in[2];
    const float* b1     = (const float*)d_in[3];
    const float* w2     = (const float*)d_in[4];
    const float* b2     = (const float*)d_in[5];
    float* out = (float*)d_out;

    // workspace layout (assumes ws_size >= ~706 MB)
    char* ws = (char*)d_ws;
    float*          contrib = (float*)ws;                 size_t off = (size_t)NT * 2 * HID * sizeof(float);
    unsigned short* w1t = (unsigned short*)(ws + off);    off += (size_t)NE * FFN * HID * 2;
    unsigned short* w2t = (unsigned short*)(ws + off);    off += (size_t)NE * HID * FFN * 2;
    unsigned short* xbf = (unsigned short*)(ws + off);    off += (size_t)NT * HID * 2;
    int*   topk_idx  = (int*)(ws + off);                  off += (size_t)NT * 2 * sizeof(int);
    float* topk_w    = (float*)(ws + off);                off += (size_t)NT * 2 * sizeof(float);
    int*   list_idx  = (int*)(ws + off);                  off += (size_t)NE * NT * sizeof(int);
    float* list_w    = (float*)(ws + off);                off += (size_t)NE * NT * sizeof(float);
    int*   list_slot = (int*)(ws + off);                  off += (size_t)NE * NT * sizeof(int);
    int*   counts    = (int*)(ws + off);                  off += 64;

    // w1 [E][H][F] -> w1t [E][F][H] ; w2 [E][F][H] -> w2t [E][H][F]
    transpose_bf16_kernel<<<NE * (HID / 32) * (FFN / 32), 256, 0, stream>>>(
        w1, w1t, HID, FFN, HID / 32, FFN / 32);
    transpose_bf16_kernel<<<NE * (FFN / 32) * (HID / 32), 256, 0, stream>>>(
        w2, w2t, FFN, HID, FFN / 32, HID / 32);
    cvt_x_kernel<<<NT * HID / 4 / 256, 256, 0, stream>>>(x, xbf);

    moe_router_kernel<<<NT / 8, 256, 0, stream>>>(x, gate_w, topk_idx, topk_w);
    moe_build_lists_kernel<<<1, 256, 0, stream>>>(topk_idx, topk_w,
                                                  list_idx, list_w, list_slot, counts);

    size_t lds_bytes = (size_t)(BM * SX_STRIDE + BM * SH_STRIDE) * sizeof(unsigned short)
                       + BM * 12;
    hipFuncSetAttribute((const void*)moe_ffn_kernel,
                        hipFuncAttributeMaxDynamicSharedMemorySize, (int)lds_bytes);
    moe_ffn_kernel<<<dim3(NE * 256), 512, lds_bytes, stream>>>(
        xbf, w1t, b1, w2t, b2, list_idx, list_w, list_slot, counts, contrib);

    moe_combine_kernel<<<(NT * HID / 4 + 255) / 256, 256, 0, stream>>>(x, contrib, out);
}
